// GKTNet_79525614453057
// MI455X (gfx1250) — matrix operand order repaired and checked by reading, not compiled, not run on hardware
//
#include <hip/hip_runtime.h>

// ---------------------------------------------------------------------------
// GKT network on gfx1250 (MI455X). wave32, WMMA f32_16x16x32_f16,
// TDM (tensor_load_to_lds) staged GEMM operands when available.
// ---------------------------------------------------------------------------

#define N_NODES 1024
#define BATCH   64
#define SEQ     64
#define SDIM    32
#define FDIM    64

typedef __attribute__((ext_vector_type(16))) _Float16 v16h;
typedef __attribute__((ext_vector_type(8)))  _Float16 v8h;
typedef __attribute__((ext_vector_type(8)))  float    v8f;

union AV { v16h v; v8h h[2]; };

__device__ __forceinline__ v8f wmma16(v16h a, v16h b, v8f c) {
  // D = A(16x32 f16) * B(32x16 f16) + C(16x16 f32)
  return __builtin_amdgcn_wmma_f32_16x16x32_f16(
      /*neg_a=*/false, a, /*neg_b=*/false, b,
      /*c_mod=*/(short)0, c, /*reuse_a=*/false, /*reuse_b=*/false);
}
__device__ __forceinline__ float sigm(float x) { return 1.0f / (1.0f + __expf(-x)); }

// ---- TDM availability ------------------------------------------------------
#if defined(__has_builtin)
#if __has_builtin(__builtin_amdgcn_tensor_load_to_lds)
#define HAVE_TDM 1
#endif
#endif
#ifndef HAVE_TDM
#define HAVE_TDM 0
#endif

__device__ __forceinline__ void wait_tensor0() {
#if defined(__has_builtin) && __has_builtin(__builtin_amdgcn_s_wait_tensorcnt)
  __builtin_amdgcn_s_wait_tensorcnt(0);
#else
  asm volatile("s_wait_tensorcnt 0x0" ::: "memory");
#endif
}

#if HAVE_TDM
typedef unsigned int u32x4 __attribute__((ext_vector_type(4)));
typedef int          i32x4 __attribute__((ext_vector_type(4)));
typedef int          i32x8 __attribute__((ext_vector_type(8)));

// 2D tile load Global->LDS via Tensor Data Mover (D# per ISA 08_async_tensor §8).
// data_size = 2 bytes; tile = tile_d0 x tile_d1 elements; row stride in elems.
__device__ __forceinline__ void tdm_load_2d(uint64_t gaddr, uint32_t lds_addr,
                                            uint32_t tile_d0, uint32_t tile_d1,
                                            uint64_t row_stride_elems,
                                            uint32_t tensor_d0, uint32_t tensor_d1) {
  u32x4 g0;
  g0.x = 1u;                                        // count=1 (valid), user mode
  g0.y = lds_addr;                                  // lds_addr (bytes)
  g0.z = (uint32_t)gaddr;                           // global_addr[31:0]
  g0.w = (uint32_t)(gaddr >> 32) | 0x80000000u;     // global_addr[56:32] | type=2
  i32x8 g1;
  g1[0] = 0x10000;                                  // wg_mask=0, data_size=1 (2B)
  g1[1] = (int)((tensor_d0 & 0xFFFFu) << 16);       // tensor_dim0[15:0]
  g1[2] = (int)((tensor_d0 >> 16) | ((tensor_d1 & 0xFFFFu) << 16));
  g1[3] = (int)((tensor_d1 >> 16) | (tile_d0 << 16));   // tile_dim0
  g1[4] = (int)(tile_d1 & 0xFFFFu);                 // tile_dim1 (tile_dim2=0)
  g1[5] = (int)(uint32_t)row_stride_elems;          // tensor_dim0_stride[31:0]
  g1[6] = (int)(uint32_t)(row_stride_elems >> 32);  // [47:32]; dim1_stride=0
  g1[7] = 0;
  i32x4 gz = {0, 0, 0, 0};
#if __clang_major__ >= 23
  i32x8 gz8 = {0, 0, 0, 0, 0, 0, 0, 0};
  __builtin_amdgcn_tensor_load_to_lds(g0, g1, gz, gz, gz8, 0);
#else
  __builtin_amdgcn_tensor_load_to_lds(g0, g1, gz, gz, 0);
#endif
}
#endif  // HAVE_TDM

// ---- workspace layout (bytes) ---------------------------------------------
static constexpr size_t OFF_H    = 0;                         //  8 MB f32 (B,N,S)
static constexpr size_t OFF_M    = (size_t)8  << 20;          // 16 MB f32 (B,N,F)
static constexpr size_t OFF_SRC  = (size_t)24 << 20;          // 16 MB f32 (B,N,F)
static constexpr size_t OFF_XP   = (size_t)40 << 20;          //  8 MB f16 packed B-op
static constexpr size_t OFF_AP   = (size_t)48 << 20;          //  2 MB f16 packed adj^T
static constexpr size_t OFF_WIN  = (size_t)50 << 20;          // 16 KB
static constexpr size_t OFF_WOUT = OFF_WIN  + 16 * 1024;      // 16 KB
static constexpr size_t OFF_WIH  = OFF_WOUT + 16 * 1024;      // 12 KB
static constexpr size_t OFF_WHH  = OFF_WIH  + 12 * 1024;      //  6 KB

// ---------------------------------------------------------------------------
// init: zero h and m carries
// ---------------------------------------------------------------------------
__global__ void k_init(float* __restrict__ h, float* __restrict__ m) {
  int stride = gridDim.x * blockDim.x;
  for (int i = blockIdx.x * blockDim.x + threadIdx.x; i < BATCH * N_NODES * SDIM; i += stride)
    h[i] = 0.0f;
  for (int i = blockIdx.x * blockDim.x + threadIdx.x; i < BATCH * N_NODES * FDIM; i += stride)
    m[i] = 0.0f;
}

// ---------------------------------------------------------------------------
// pack adj^T into WMMA A-layout f16 tiles (step invariant).
// A tile (16M x 32K): lane<16 -> M=lane, K = kt*32 + {0..7,16..23}
//                     lane>=16 -> M=lane-16, K = kt*32 + {8..15,24..31}
// ---------------------------------------------------------------------------
__global__ void k_packA(const float* __restrict__ adj, _Float16* __restrict__ Ap) {
  int tid  = blockIdx.x * blockDim.x + threadIdx.x;   // 64*32*32 = 65536
  int lane = tid & 31;
  int kt   = (tid >> 5) & 31;
  int mt   = tid >> 10;
  int row  = lane & 15;
  int kb   = (lane < 16) ? 0 : 8;
  int Mg   = mt * 16 + row;
  _Float16* o = Ap + ((size_t)(mt * 32 + kt) * 32 + lane) * 16;
#pragma unroll
  for (int i = 0; i < 8; ++i) {
    int K0 = kt * 32 + kb + i;
    int K1 = K0 + 16;
    // A[v][u] = adjT[v][u] = adj[u][v]
    o[i]     = (_Float16)adj[(size_t)K0 * N_NODES + Mg];
    o[8 + i] = (_Float16)adj[(size_t)K1 * N_NODES + Mg];
  }
}

// ---------------------------------------------------------------------------
// pack weight matrices into WMMA B-layout f16 tiles (step invariant).
// B tile (32K x 16N): lane<16 -> N=lane, K = kt*32 + {0..7,16..23}
//                     lane>=16 -> N=lane-16, K = kt*32 + {8..15,24..31}
// B[k][n] = W[n][k]  (because y = x @ W^T)
// ---------------------------------------------------------------------------
__device__ __forceinline__ void packB_one(const float* __restrict__ W, int Kdim,
                                          int nt, int kt, int lane,
                                          _Float16* __restrict__ o) {
  int N  = lane & 15;
  int kb = (lane < 16) ? 0 : 8;
  const float* wr = W + ((size_t)(nt * 16 + N)) * Kdim + kt * 32 + kb;
#pragma unroll
  for (int i = 0; i < 16; ++i) o[i] = (_Float16)wr[i + (i >= 8 ? 8 : 0)];
}

__global__ void k_packW(const float* __restrict__ w_in, const float* __restrict__ w_out,
                        const float* __restrict__ w_ih, const float* __restrict__ w_hh,
                        _Float16* __restrict__ WinP, _Float16* __restrict__ WoutP,
                        _Float16* __restrict__ WihP, _Float16* __restrict__ WhhP) {
  int tid  = blockIdx.x * blockDim.x + threadIdx.x;
  int lane = tid & 31;
  int tile = tid >> 5;
  if (tile < 16) {                       // w_in : (64,128) -> 4 nt x 4 kt
    int nt = tile >> 2, kt = tile & 3;
    packB_one(w_in, 128, nt, kt, lane, WinP + ((size_t)tile * 32 + lane) * 16);
  } else if (tile < 32) {                // w_out
    int tt = tile - 16, nt = tt >> 2, kt = tt & 3;
    packB_one(w_out, 128, nt, kt, lane, WoutP + ((size_t)tt * 32 + lane) * 16);
  } else if (tile < 44) {                // w_ih : (96,64) -> 6 nt x 2 kt
    int tt = tile - 32, nt = tt >> 1, kt = tt & 1;
    packB_one(w_ih, 64, nt, kt, lane, WihP + ((size_t)tt * 32 + lane) * 16);
  } else if (tile < 50) {                // w_hh : (96,32) -> 6 nt x 1 kt
    int tt = tile - 44;
    packB_one(w_hh, 32, tt, 0, lane, WhhP + ((size_t)tt * 32 + lane) * 16);
  }
}

// ---------------------------------------------------------------------------
// per-step: build B-operand Xp[u, col]  (col = b*64+f), f16 B-layout.
// X[u][col] = f<32 ? h[b,u,f] : merged[b,u,f-32]
// ---------------------------------------------------------------------------
__global__ void k_prepare(int t, const int* __restrict__ item_ids,
                          const int* __restrict__ responses,
                          const float* __restrict__ h,
                          const float* __restrict__ item_emb,
                          const float* __restrict__ resp_emb,
                          _Float16* __restrict__ Xp) {
  int tid   = blockIdx.x * blockDim.x + threadIdx.x;   // 256*32*32 = 262144
  int lane  = tid & 31;
  int kt    = (tid >> 5) & 31;
  int ntile = tid >> 10;                               // 0..255
  int col   = ntile * 16 + (lane & 15);
  int b = col >> 6, f = col & 63;
  int kb    = (lane < 16) ? 0 : 8;
  int item  = item_ids[b * SEQ + t];
  int resp  = responses[b * SEQ + t];
  _Float16* o = Xp + ((size_t)(ntile * 32 + kt) * 32 + lane) * 16;
#pragma unroll
  for (int i = 0; i < 16; ++i) {
    int u = kt * 32 + kb + i + (i >= 8 ? 8 : 0);
    float v;
    if (f < SDIM) {
      v = h[((size_t)b * N_NODES + u) * SDIM + f];
    } else {
      int fd = f - SDIM;
      v = (u == item) ? resp_emb[(size_t)resp * SDIM + fd]
                      : item_emb[(size_t)u * SDIM + fd];
    }
    o[i] = (_Float16)v;
  }
}

// ---------------------------------------------------------------------------
// big GEMM: src(1024 x 4096) = adjT(1024x1024) * X(1024x4096), f16 in/f32 out.
// LDS-blocked: block = 8 M-tiles x 8 N-tiles, 16 waves (512 thr).
// Per K-step the block stages 8 A-tiles (8KB) + 8 B-tiles (8KB) into LDS,
// double-buffered. With TDM, wave 0 issues two tensor_load_to_lds DMAs
// (512x8 f16 tiles, row stride 16384 elems) and fences with s_wait_tensorcnt;
// otherwise threads stage one 32B packed entry each (branchless ptr select).
// Each wave computes 1 M-tile x 4 N-tiles from LDS.
// ---------------------------------------------------------------------------
__global__ void __launch_bounds__(512) k_gemm(const _Float16* __restrict__ Ap,
                                              const _Float16* __restrict__ Xp,
                                              float* __restrict__ src) {
  __shared__ v16h As[2][8 * 32];
  __shared__ v16h Bs[2][8 * 32];

  const int tid  = threadIdx.x;
  const int lane = tid & 31;
  const int wv   = tid >> 5;                 // 0..15
  const int mb   = blockIdx.x & 7;           // m-block: mtiles mb*8 .. +7
  const int nb   = blockIdx.x >> 3;          // n-block: ntiles nb*8 .. +7 (0..31)
  const int mt0  = mb * 8;
  const int nt0  = nb * 8;

  const int mtl = wv & 7;                    // wave's local m-tile
  const int ngh = wv >> 3;                   // wave's n-half (0/1) -> 4 ntiles
  v8f c0 = {}, c1 = {}, c2 = {}, c3 = {};
  int buf = 0;

#if HAVE_TDM
  // tile (mt,kt) starts at byte (mt*32+kt)*1024; row (tl) stride = 32 KB.
  const uint64_t gA = (uint64_t)(uintptr_t)Ap + (uint64_t)mt0 * 32u * 1024u;
  const uint64_t gB = (uint64_t)(uintptr_t)Xp + (uint64_t)nt0 * 32u * 1024u;
  if (wv == 0) {
    tdm_load_2d(gA, (uint32_t)(uintptr_t)&As[0][0], 512u, 8u, 16384u, 16384u, 8u);
    tdm_load_2d(gB, (uint32_t)(uintptr_t)&Bs[0][0], 512u, 8u, 16384u, 16384u, 8u);
  }
  for (int kt = 0; kt < 32; ++kt) {
    if (wv == 0) wait_tensor0();             // buf fully DMA'd into LDS
    __syncthreads();                         // release readers / writers of buf^1
    if (wv == 0 && kt + 1 < 32) {
      tdm_load_2d(gA + (uint64_t)(kt + 1) * 1024u,
                  (uint32_t)(uintptr_t)&As[buf ^ 1][0], 512u, 8u, 16384u, 16384u, 8u);
      tdm_load_2d(gB + (uint64_t)(kt + 1) * 1024u,
                  (uint32_t)(uintptr_t)&Bs[buf ^ 1][0], 512u, 8u, 16384u, 16384u, 8u);
    }
    v16h a  = As[buf][mtl * 32 + lane];
    v16h b0 = Bs[buf][(ngh * 4 + 0) * 32 + lane];
    v16h b1 = Bs[buf][(ngh * 4 + 1) * 32 + lane];
    v16h b2 = Bs[buf][(ngh * 4 + 2) * 32 + lane];
    v16h b3 = Bs[buf][(ngh * 4 + 3) * 32 + lane];
    c0 = wmma16(a, b0, c0);
    c1 = wmma16(a, b1, c1);
    c2 = wmma16(a, b2, c2);
    c3 = wmma16(a, b3, c3);
    buf ^= 1;
  }
#else
  // manual staging: threads 0..255 -> A entries, 256..511 -> B entries
  const v16h* Ag = (const v16h*)Ap;
  const v16h* Bg = (const v16h*)Xp;
  const int  sIdx  = tid & 255;
  const int  sTile = sIdx >> 5;
  const int  sLane = sIdx & 31;
  const bool isA   = tid < 256;
  const v16h* gsrc = isA ? (Ag + ((size_t)(mt0 + sTile) * 32) * 32 + sLane)
                         : (Bg + ((size_t)(nt0 + sTile) * 32) * 32 + sLane);
  v16h* d0 = isA ? &As[0][sIdx] : &Bs[0][sIdx];     // branchless dst select
  v16h* d1 = isA ? &As[1][sIdx] : &Bs[1][sIdx];
  *d0 = gsrc[0];                                    // stage kt = 0
  for (int kt = 0; kt < 32; ++kt) {
    __syncthreads();                                // buf staged, buf^1 free
    v16h a  = As[buf][mtl * 32 + lane];
    v16h b0 = Bs[buf][(ngh * 4 + 0) * 32 + lane];
    v16h b1 = Bs[buf][(ngh * 4 + 1) * 32 + lane];
    v16h b2 = Bs[buf][(ngh * 4 + 2) * 32 + lane];
    v16h b3 = Bs[buf][(ngh * 4 + 3) * 32 + lane];
    if (kt + 1 < 32) {
      v16h g = gsrc[(kt + 1) * 32];
      if (kt + 2 < 32)
        __builtin_prefetch((const void*)(gsrc + (kt + 2) * 32), 0, 1);
      *((buf == 0) ? d1 : d0) = g;
    }
    c0 = wmma16(a, b0, c0);
    c1 = wmma16(a, b1, c1);
    c2 = wmma16(a, b2, c2);
    c3 = wmma16(a, b3, c3);
    buf ^= 1;
  }
#endif

  // C layout: VGPR r, lane<16 -> (M=r, N=lane); lane>=16 -> (M=r+8, N=lane-16)
  const int vb = (mt0 + mtl) * 16 + ((lane >= 16) ? 8 : 0);
  v8f cs[4] = {c0, c1, c2, c3};
#pragma unroll
  for (int j = 0; j < 4; ++j) {
    int col = (nt0 + ngh * 4 + j) * 16 + (lane & 15);
    int b = col >> 6, f = col & 63;
    float* sp = src + ((size_t)b * N_NODES + vb) * FDIM + f;
#pragma unroll
    for (int r = 0; r < 8; ++r) sp[(size_t)r * FDIM] = cs[j][r];
  }
}

// ---------------------------------------------------------------------------
// fused1: fea=[h, merged, src] (16 rows x 128) staged in LDS ->
//         in_fea/out_fea via WMMA -> dst = out_a*out + in_a*in ->
//         m_new = neigh ? dst : m  (in place on m)
// 1 wave per 16 consecutive (b, v) rows; block = 4 waves.
// ---------------------------------------------------------------------------
__global__ void __launch_bounds__(128) k_fused1(
    int t, const int* __restrict__ item_ids, const int* __restrict__ responses,
    const float* __restrict__ adj, const float* __restrict__ h,
    const float* __restrict__ src, const float* __restrict__ item_emb,
    const float* __restrict__ resp_emb,
    const _Float16* __restrict__ WinP, const _Float16* __restrict__ WoutP,
    const float* __restrict__ b_in, const float* __restrict__ b_out,
    const float* __restrict__ in_a, const float* __restrict__ out_a,
    float* __restrict__ m) {
  constexpr int STR = 136;                       // 128 + 8 f16 pad (bank skew)
  __shared__ _Float16 sm[4][16 * STR];
  int w    = threadIdx.x >> 5;
  int lane = threadIdx.x & 31;
  int grp  = blockIdx.x * 4 + w;                 // 0..4095
  int b    = grp >> 6;
  int v0   = (grp & 63) << 4;
  int item = item_ids[b * SEQ + t];
  int resp = responses[b * SEQ + t];

  // stage fea rows: lane<16 -> [h | merged], lane>=16 -> [src]
  int row = lane & 15, half = lane >> 4;
  int v = v0 + row;
  _Float16* srow = sm[w] + row * STR;
  if (half == 0) {
    const float* hp = h + ((size_t)b * N_NODES + v) * SDIM;
#pragma unroll
    for (int j = 0; j < SDIM; ++j) srow[j] = (_Float16)hp[j];
    const float* ep = (v == item) ? (resp_emb + (size_t)resp * SDIM)
                                  : (item_emb + (size_t)v * SDIM);
#pragma unroll
    for (int j = 0; j < SDIM; ++j) srow[SDIM + j] = (_Float16)ep[j];
  } else {
    const float* sp = src + ((size_t)b * N_NODES + v) * FDIM;
#pragma unroll
    for (int j = 0; j < FDIM; ++j) srow[FDIM + j] = (_Float16)sp[j];
  }
  __syncthreads();

  // A-tiles (16x32 f16) from LDS, ISA lane layout
  AV a[4];
  int abase = (lane < 16) ? 0 : 8;
  const _Float16* ar = sm[w] + (lane & 15) * STR;
#pragma unroll
  for (int kt = 0; kt < 4; ++kt) {
    a[kt].h[0] = *(const v8h*)(ar + kt * 32 + abase);
    a[kt].h[1] = *(const v8h*)(ar + kt * 32 + abase + 16);
  }

  const v16h* Bi = (const v16h*)WinP + lane;
  const v16h* Bo = (const v16h*)WoutP + lane;

  int vb = v0 + ((lane >= 16) ? 8 : 0);
  float ina[8], outa[8];
  bool  ng[8];
#pragma unroll
  for (int r = 0; r < 8; ++r) {
    int vr = vb + r;
    ina[r]  = in_a[vr];
    outa[r] = out_a[vr];
    ng[r]   = adj[(size_t)vr * N_NODES + item] > 0.0f;   // in-neighborhood of item
  }

#pragma unroll
  for (int nt = 0; nt < 4; ++nt) {
    v8f ci = {}, co = {};
#pragma unroll
    for (int kt = 0; kt < 4; ++kt) {
      ci = wmma16(a[kt].v, Bi[(size_t)(nt * 4 + kt) * 32], ci);
      co = wmma16(a[kt].v, Bo[(size_t)(nt * 4 + kt) * 32], co);
    }
    int f = nt * 16 + (lane & 15);
    float bi = b_in[f], bo = b_out[f];
    float* mp = m + ((size_t)b * N_NODES + vb) * FDIM + f;
#pragma unroll
    for (int r = 0; r < 8; ++r) {
      float dst = outa[r] * (co[r] + bo) + ina[r] * (ci[r] + bi);
      float mo  = mp[(size_t)r * FDIM];
      mp[(size_t)r * FDIM] = ng[r] ? dst : mo;
    }
  }
}

// ---------------------------------------------------------------------------
// fused2: GRU over 16 rows: gi = x@w_ih^T, gh = h@w_hh^T via WMMA,
// gates elementwise, h update in place, out_t = sigmoid(h_new @ w_fc^T).
// ---------------------------------------------------------------------------
__global__ void __launch_bounds__(128) k_fused2(
    int t, int last, const float* __restrict__ m,
    const _Float16* __restrict__ WihP, const _Float16* __restrict__ WhhP,
    const float* __restrict__ b_ih, const float* __restrict__ b_hh,
    const float* __restrict__ w_fc, const float* __restrict__ b_fc,
    float* __restrict__ h, float* __restrict__ out) {
  constexpr int XS = 72, HS = 40;                // padded f16 row strides
  __shared__ _Float16 sx[4][16 * XS];
  __shared__ _Float16 shh[4][16 * HS];
  __shared__ float    sn[4][16][33];
  int w    = threadIdx.x >> 5;
  int lane = threadIdx.x & 31;
  int grp  = blockIdx.x * 4 + w;
  int b    = grp >> 6;
  int v0   = (grp & 63) << 4;
  int row  = lane & 15, half = lane >> 4;
  int v    = v0 + row;

  {  // stage x = m_new rows and hp rows as f16
    const float* mp = m + ((size_t)b * N_NODES + v) * FDIM + half * 32;
    _Float16* xr = sx[w] + row * XS + half * 32;
#pragma unroll
    for (int j = 0; j < 32; ++j) xr[j] = (_Float16)mp[j];
    const float* hp = h + ((size_t)b * N_NODES + v) * SDIM + half * 16;
    _Float16* hr = shh[w] + row * HS + half * 16;
#pragma unroll
    for (int j = 0; j < 16; ++j) hr[j] = (_Float16)hp[j];
  }
  __syncthreads();

  AV ax[2], ah;
  int abase = (lane < 16) ? 0 : 8;
  const _Float16* xr = sx[w] + (lane & 15) * XS;
#pragma unroll
  for (int kt = 0; kt < 2; ++kt) {
    ax[kt].h[0] = *(const v8h*)(xr + kt * 32 + abase);
    ax[kt].h[1] = *(const v8h*)(xr + kt * 32 + abase + 16);
  }
  const _Float16* hr = shh[w] + (lane & 15) * HS;
  ah.h[0] = *(const v8h*)(hr + abase);
  ah.h[1] = *(const v8h*)(hr + abase + 16);

  const v16h* Bih = (const v16h*)WihP + lane;
  const v16h* Bhh = (const v16h*)WhhP + lane;
  v8f gi[6], gh[6];
#pragma unroll
  for (int nt = 0; nt < 6; ++nt) {
    v8f c = {};
    c = wmma16(ax[0].v, Bih[(size_t)(nt * 2 + 0) * 32], c);
    c = wmma16(ax[1].v, Bih[(size_t)(nt * 2 + 1) * 32], c);
    gi[nt] = c;
    v8f d = {};
    d = wmma16(ah.v, Bhh[(size_t)nt * 32], d);
    gh[nt] = d;
  }

  int nn = lane & 15;
  int vb = v0 + ((lane >= 16) ? 8 : 0);
  int mb = (lane >= 16) ? 8 : 0;
#pragma unroll
  for (int j = 0; j < 2; ++j) {                       // state cols j*16 .. +15
    int colr = j * 16 + nn;
    float bir = b_ih[colr],      bhr = b_hh[colr];
    float biz = b_ih[32 + colr], bhz = b_hh[32 + colr];
    float bin = b_ih[64 + colr], bhn = b_hh[64 + colr];
    float* hptr = h + ((size_t)b * N_NODES + vb) * SDIM + colr;
#pragma unroll
    for (int r = 0; r < 8; ++r) {
      float hpv = hptr[(size_t)r * SDIM];
      float rr = sigm(gi[j][r] + bir + gh[j][r] + bhr);
      float zz = sigm(gi[2 + j][r] + biz + gh[2 + j][r] + bhz);
      float nv = tanhf(gi[4 + j][r] + bin + rr * (gh[4 + j][r] + bhn));
      float hn = (1.0f - zz) * nv + zz * hpv;
      hptr[(size_t)r * SDIM] = hn;
      sn[w][mb + r][colr] = hn;
    }
  }
  __syncthreads();

  if (lane < 16) {
    float acc = b_fc[0];
#pragma unroll
    for (int s = 0; s < SDIM; ++s) acc += sn[w][lane][s] * w_fc[s];
    out[((size_t)b * SEQ + t) * N_NODES + v0 + lane] = sigm(acc);
    if (last) {
      float* ho = out + (size_t)BATCH * SEQ * N_NODES +
                  ((size_t)b * N_NODES + v0 + lane) * SDIM;
#pragma unroll
      for (int s = 0; s < SDIM; ++s) ho[s] = sn[w][lane][s];
    }
  }
}

// ---------------------------------------------------------------------------
extern "C" void kernel_launch(void* const* d_in, const int* in_sizes, int n_in,
                              void* d_out, int out_size, void* d_ws, size_t ws_size,
                              hipStream_t stream) {
  (void)in_sizes; (void)n_in; (void)out_size; (void)ws_size;
  const int*   item_ids  = (const int*)d_in[0];
  const int*   responses = (const int*)d_in[1];
  const float* adj       = (const float*)d_in[2];
  const float* item_emb  = (const float*)d_in[3];
  const float* resp_emb  = (const float*)d_in[4];
  const float* w_in      = (const float*)d_in[5];
  const float* b_in      = (const float*)d_in[6];
  const float* w_out     = (const float*)d_in[7];
  const float* b_out     = (const float*)d_in[8];
  const float* in_a      = (const float*)d_in[9];
  const float* out_a     = (const float*)d_in[10];
  const float* w_ih      = (const float*)d_in[11];
  const float* w_hh      = (const float*)d_in[12];
  const float* b_ih      = (const float*)d_in[13];
  const float* b_hh      = (const float*)d_in[14];
  const float* w_fc      = (const float*)d_in[15];
  const float* b_fc      = (const float*)d_in[16];

  char* ws = (char*)d_ws;
  float*     h    = (float*)(ws + OFF_H);
  float*     mbuf = (float*)(ws + OFF_M);
  float*     src  = (float*)(ws + OFF_SRC);
  _Float16*  Xp   = (_Float16*)(ws + OFF_XP);
  _Float16*  Ap   = (_Float16*)(ws + OFF_AP);
  _Float16*  WinP = (_Float16*)(ws + OFF_WIN);
  _Float16*  WoutP= (_Float16*)(ws + OFF_WOUT);
  _Float16*  WihP = (_Float16*)(ws + OFF_WIH);
  _Float16*  WhhP = (_Float16*)(ws + OFF_WHH);
  float* out = (float*)d_out;

  k_init <<<2048, 256, 0, stream>>>(h, mbuf);
  k_packA<<<256, 256, 0, stream>>>(adj, Ap);
  k_packW<<<7, 256, 0, stream>>>(w_in, w_out, w_ih, w_hh, WinP, WoutP, WihP, WhhP);

  for (int t = 0; t < SEQ; ++t) {
    k_prepare<<<1024, 256, 0, stream>>>(t, item_ids, responses, h, item_emb, resp_emb, Xp);
    k_gemm   <<<256, 512, 0, stream>>>(Ap, Xp, src);
    k_fused1 <<<1024, 128, 0, stream>>>(t, item_ids, responses, adj, h, src,
                                        item_emb, resp_emb, WinP, WoutP,
                                        b_in, b_out, in_a, out_a, mbuf);
    k_fused2 <<<1024, 128, 0, stream>>>(t, (t == SEQ - 1) ? 1 : 0, mbuf,
                                        WihP, WhhP, b_ih, b_hh, w_fc, b_fc, h, out);
  }
}
